// LinearTransformer_26225070309969
// MI455X (gfx1250) — matrix operand order repaired and checked by reading, not compiled, not run on hardware
//
#include <hip/hip_runtime.h>
#include <math.h>

// ---------------------------------------------------------------------------
// Types
// ---------------------------------------------------------------------------
typedef __bf16 bf16_t;
typedef __bf16 bf16x8  __attribute__((ext_vector_type(8)));
typedef __bf16 bf16x16 __attribute__((ext_vector_type(16)));
typedef float  v8f     __attribute__((ext_vector_type(8)));
typedef int    v4i     __attribute__((ext_vector_type(4)));

union Frag16 { bf16x16 v; bf16x8 h[2]; };

#define AS1 __attribute__((address_space(1)))
#define AS3 __attribute__((address_space(3)))

// CDNA5 async global->LDS path (ASYNCcnt-tracked), guarded so compile never breaks.
#if defined(__AMDGCN__) && __has_builtin(__builtin_amdgcn_global_load_async_to_lds_b128)
#define HAVE_ASYNC_LDS 1
#else
#define HAVE_ASYNC_LDS 0
#endif

#if defined(__AMDGCN__) && __has_builtin(__builtin_amdgcn_s_wait_asynccnt)
#define WAIT_ASYNC(n) __builtin_amdgcn_s_wait_asynccnt(n)
#elif defined(__AMDGCN__)
#define WAIT_ASYNC(n) asm volatile("s_wait_asynccnt %0" ::"i"(n) : "memory")
#else
#define WAIT_ASYNC(n)
#endif

__device__ __forceinline__ bf16_t f2bf(float f) {
    union { float f; unsigned u; } c; c.f = f;
    unsigned r = (c.u + 0x7FFFu + ((c.u >> 16) & 1u)) >> 16;
    union { unsigned short s; bf16_t b; } o; o.s = (unsigned short)r;
    return o.b;
}

__device__ __forceinline__ float gelu_tanh(float z) {
    // jax.nn.gelu (approximate=True)
    float z3 = z * z * z;
    return 0.5f * z * (1.0f + tanhf(0.7978845608028654f * (z + 0.044715f * z3)));
}

// ---------------------------------------------------------------------------
// Problem constants
// ---------------------------------------------------------------------------
#define NB 4
#define LL 4096
#define DD 1024
#define AA 1024
#define FF 4096
#define BBLK 8
#define MM (NB * LL)   // 16384 rows

#define EPS_LN 1e-6f
#define EPS_Z  1e-6f

// GEMM tiling
#define TM 128
#define TN 128
#define KT 32
#define LDSP 56        // bf16 elements of LDS row pitch (112B, 16B-aligned, conflict-light)

// ---------------------------------------------------------------------------
// Elementwise / helper kernels
// ---------------------------------------------------------------------------
__global__ void convert_x_kernel(const float* __restrict__ x,
                                 float* __restrict__ xf, bf16_t* __restrict__ xb, int n) {
    int i = blockIdx.x * 256 + threadIdx.x;
    if (i < n) { float v = x[i]; xf[i] = v; xb[i] = f2bf(v); }
}

// W[K][Nn] fp32 (row-major) -> Wt[Nn][K] bf16 (row-major), LDS-tiled 32x32 so
// both the fp32 reads and the bf16 writes are contiguous 64B+ segments.
__global__ __launch_bounds__(256)
void transpose_w_kernel(const float* __restrict__ W, bf16_t* __restrict__ Wt,
                        int K, int Nn) {
    __shared__ bf16_t tile[32][33];
    const int tx = threadIdx.x & 31;   // fast dim
    const int ty = threadIdx.x >> 5;   // 8 rows per pass
    const int n0 = blockIdx.x * 32;
    const int k0 = blockIdx.y * 32;
    for (int p = 0; p < 32; p += 8)
        tile[ty + p][tx] = f2bf(W[(size_t)(k0 + ty + p) * Nn + n0 + tx]);
    __syncthreads();
    for (int p = 0; p < 32; p += 8)
        Wt[(size_t)(n0 + ty + p) * K + k0 + tx] = tile[tx][ty + p];
}

__global__ void zero_kernel(float* __restrict__ p, int n) {
    int i = blockIdx.x * 256 + threadIdx.x;
    if (i < n) p[i] = 0.0f;
}

// KV[n,a] = sum_l K[n,l,a]*v[n,l,a] ; Ksum[n,a] = sum_l K[n,l,a]
// grid.x = NB*(AA/256), grid.y = 16 chunks over L
__global__ void kv_reduce_kernel(const float* __restrict__ Kf, const float* __restrict__ vf,
                                 float* __restrict__ KV, float* __restrict__ Ksum) {
    int a = (blockIdx.x % (AA / 256)) * 256 + threadIdx.x;
    int n = blockIdx.x / (AA / 256);
    int l0 = blockIdx.y * (LL / 16);
    float skv = 0.0f, sk = 0.0f;
    size_t base = ((size_t)n * LL + l0) * AA + a;
    for (int l = 0; l < LL / 16; ++l) {
        float kk = Kf[base], vv = vf[base];
        skv += kk * vv; sk += kk;
        base += AA;
    }
    atomicAdd(&KV[n * AA + a], skv);
    atomicAdd(&Ksum[n * AA + a], sk);
}

// V = Q * KV / (Q * Ksum + eps)  -> bf16
__global__ void v_map_kernel(const float* __restrict__ Qf, const float* __restrict__ KV,
                             const float* __restrict__ Ksum, bf16_t* __restrict__ Vb) {
    size_t i = (size_t)blockIdx.x * 256 + threadIdx.x;
    int a = (int)(i % AA);
    int n = (int)(i / ((size_t)LL * AA));
    float q = Qf[i];
    float z = 1.0f / (q * Ksum[n * AA + a] + EPS_Z);
    Vb[i] = f2bf(q * KV[n * AA + a] * z);
}

// LayerNorm over D=1024 per row; one 256-thread block per row.
__global__ __launch_bounds__(256)
void ln_kernel(const float* __restrict__ y, const float* __restrict__ s,
               const float* __restrict__ b, float* __restrict__ dstF,
               bf16_t* __restrict__ dstH) {
    __shared__ float r1[256];
    __shared__ float r2[256];
    int row = blockIdx.x, tid = threadIdx.x;
    size_t base = (size_t)row * DD;
    float xv[4], sa = 0.0f, sb = 0.0f;
    for (int p = 0; p < 4; ++p) {
        float v = y[base + tid + p * 256];
        xv[p] = v; sa += v; sb += v * v;
    }
    r1[tid] = sa; r2[tid] = sb;
    __syncthreads();
    for (int off = 128; off > 0; off >>= 1) {
        if (tid < off) { r1[tid] += r1[tid + off]; r2[tid] += r2[tid + off]; }
        __syncthreads();
    }
    float mean = r1[0] * (1.0f / DD);
    float var  = r2[0] * (1.0f / DD) - mean * mean;
    float rstd = rsqrtf(var + EPS_LN);
    for (int p = 0; p < 4; ++p) {
        int e = tid + p * 256;
        float o = (xv[p] - mean) * rstd * s[e] + b[e];
        dstF[base + e] = o;
        if (dstH) dstH[base + e] = f2bf(o);
    }
}

// ---------------------------------------------------------------------------
// WMMA bf16 GEMM:  out = epilogue( A[M,K] * Bt[Nn,K]^T + bias )
//   EPI 0: raw f32        EPI 1: elu(z)+1 f32      EPI 2: bias-only f32
//   EPI 3: gelu -> bf16   EPI 4: gelu + residual -> f32
// Async double-buffered LDS pipeline when GLOBAL_LOAD_ASYNC_TO_LDS is available.
// ---------------------------------------------------------------------------
template <int EPI>
__global__ __launch_bounds__(256)
void gemm_bf16_wmma(const bf16_t* __restrict__ Amat, const bf16_t* __restrict__ Bt,
                    const float* __restrict__ bias, const float* __restrict__ resid,
                    float* __restrict__ outF, bf16_t* __restrict__ outH,
                    int M, int Nn, int K) {
#if HAVE_ASYNC_LDS
    __shared__ bf16_t lA[2][TM * LDSP];
    __shared__ bf16_t lB[2][TN * LDSP];
#else
    __shared__ bf16_t lA[1][TM * LDSP];
    __shared__ bf16_t lB[1][TN * LDSP];
#endif

    const int tid  = threadIdx.x;
    const int lane = tid & 31;
    const int wid  = tid >> 5;
    const int mw   = wid & 1;    // 2 wave-rows
    const int nw   = wid >> 1;   // 4 wave-cols
    const int m0   = blockIdx.y * TM;
    const int n0   = blockIdx.x * TN;

    const int lr = tid >> 2;     // 0..63 : tile row for global->LDS
    const int lc = tid & 3;      // 16B chunk within 64B row

    const int mf   = lane & 15;
    const int half = lane >> 4;

    v8f acc[4][2];
    const v8f vzero = {0.f, 0.f, 0.f, 0.f, 0.f, 0.f, 0.f, 0.f};
    for (int i = 0; i < 4; ++i)
        for (int j = 0; j < 2; ++j) acc[i][j] = vzero;

    Frag16 af[4], bfg[2];

#if HAVE_ASYNC_LDS
    // ---- async double-buffered pipeline: 4 x b128 async ops per lane per tile
    auto issue_tile = [&](int k0, int buf) {
        bf16_t* la = lA[buf];
        bf16_t* lb = lB[buf];
        for (int rr = 0; rr < TM; rr += 64) {
            __builtin_amdgcn_global_load_async_to_lds_b128(
                (AS1 v4i*)(Amat + (size_t)(m0 + rr + lr) * K + k0 + lc * 8),
                (AS3 v4i*)&la[(rr + lr) * LDSP + lc * 8], 0, 0);
            __builtin_amdgcn_global_load_async_to_lds_b128(
                (AS1 v4i*)(Bt + (size_t)(n0 + rr + lr) * K + k0 + lc * 8),
                (AS3 v4i*)&lb[(rr + lr) * LDSP + lc * 8], 0, 0);
        }
    };

    const int T = K / KT;
    issue_tile(0, 0);
    for (int t = 0; t < T; ++t) {
        const int cur = t & 1;
        if (t + 1 < T) {
            issue_tile((t + 1) * KT, 1 - cur);  // safe: end-barrier of t-1 passed
            WAIT_ASYNC(4);                      // tile t landed, prefetch in flight
        } else {
            WAIT_ASYNC(0);
        }
        __syncthreads();                        // tile t visible to all waves

        const bf16_t* la = lA[cur];
        const bf16_t* lb = lB[cur];
        for (int i = 0; i < 4; ++i) {
            int row = mw * 64 + i * 16 + mf;
            af[i].h[0] = *(const bf16x8*)&la[row * LDSP + half * 8];
            af[i].h[1] = *(const bf16x8*)&la[row * LDSP + 16 + half * 8];
        }
        for (int j = 0; j < 2; ++j) {
            int col = nw * 32 + j * 16 + mf;
            bfg[j].h[0] = *(const bf16x8*)&lb[col * LDSP + half * 8];
            bfg[j].h[1] = *(const bf16x8*)&lb[col * LDSP + 16 + half * 8];
        }
        for (int i = 0; i < 4; ++i)
            for (int j = 0; j < 2; ++j)
                acc[i][j] = __builtin_amdgcn_wmma_f32_16x16x32_bf16(
                    false, af[i].v, false, bfg[j].v, (short)0, acc[i][j],
                    false, false);
        __syncthreads();                        // reads done before buffer reuse
    }
#else
    // ---- fallback: synchronous global->VGPR->LDS staging
    for (int k0 = 0; k0 < K; k0 += KT) {
        __syncthreads();
        for (int rr = 0; rr < TM; rr += 64) {
            bf16x8 va = *(const bf16x8*)(Amat + (size_t)(m0 + rr + lr) * K + k0 + lc * 8);
            *(bf16x8*)&lA[0][(rr + lr) * LDSP + lc * 8] = va;
            bf16x8 vb = *(const bf16x8*)(Bt + (size_t)(n0 + rr + lr) * K + k0 + lc * 8);
            *(bf16x8*)&lB[0][(rr + lr) * LDSP + lc * 8] = vb;
        }
        __syncthreads();
        for (int i = 0; i < 4; ++i) {
            int row = mw * 64 + i * 16 + mf;
            af[i].h[0] = *(const bf16x8*)&lA[0][row * LDSP + half * 8];
            af[i].h[1] = *(const bf16x8*)&lA[0][row * LDSP + 16 + half * 8];
        }
        for (int j = 0; j < 2; ++j) {
            int col = nw * 32 + j * 16 + mf;
            bfg[j].h[0] = *(const bf16x8*)&lB[0][col * LDSP + half * 8];
            bfg[j].h[1] = *(const bf16x8*)&lB[0][col * LDSP + 16 + half * 8];
        }
        for (int i = 0; i < 4; ++i)
            for (int j = 0; j < 2; ++j)
                acc[i][j] = __builtin_amdgcn_wmma_f32_16x16x32_bf16(
                    false, af[i].v, false, bfg[j].v, (short)0, acc[i][j],
                    false, false);
    }
#endif

    // Epilogue: D-tile layout — lane holds column (lane&15), rows r + 8*(lane>>4)
    for (int i = 0; i < 4; ++i) {
        int mrow0 = m0 + mw * 64 + i * 16 + half * 8;
        for (int j = 0; j < 2; ++j) {
            int ncol = n0 + nw * 32 + j * 16 + mf;
            float bv = (EPI == 0) ? 0.0f : bias[ncol];
            for (int r = 0; r < 8; ++r) {
                size_t idx = (size_t)(mrow0 + r) * Nn + ncol;
                float z = acc[i][j][r] + bv;
                if (EPI == 1) {                       // elu(z) + 1
                    z = (z > 0.0f) ? (z + 1.0f) : __expf(z);
                } else if (EPI == 3) {                // gelu -> bf16
                    outH[idx] = f2bf(gelu_tanh(z));
                    continue;
                } else if (EPI == 4) {                // gelu + residual -> f32
                    z = gelu_tanh(z) + resid[idx];
                }
                outF[idx] = z;
            }
        }
    }
}

// ---------------------------------------------------------------------------
// Host orchestration
// ---------------------------------------------------------------------------
extern "C" void kernel_launch(void* const* d_in, const int* in_sizes, int n_in,
                              void* d_out, int out_size, void* d_ws, size_t ws_size,
                              hipStream_t stream) {
    const float* x    = (const float*)d_in[0];
    const float* Wq   = (const float*)d_in[1];
    const float* bq   = (const float*)d_in[2];
    const float* Wk   = (const float*)d_in[3];
    const float* bk   = (const float*)d_in[4];
    const float* Wv   = (const float*)d_in[5];
    const float* bv   = (const float*)d_in[6];
    const float* Wo   = (const float*)d_in[7];
    const float* bo   = (const float*)d_in[8];
    const float* ln1s = (const float*)d_in[9];
    const float* ln1b = (const float*)d_in[10];
    const float* Wf1  = (const float*)d_in[11];
    const float* bf1  = (const float*)d_in[12];
    const float* Wf2  = (const float*)d_in[13];
    const float* bf2  = (const float*)d_in[14];
    const float* ln2s = (const float*)d_in[15];
    const float* ln2b = (const float*)d_in[16];
    float* out = (float*)d_out;

    // Workspace carve-up
    char* ws = (char*)d_ws;
    size_t off = 0;
    auto carve = [&](size_t bytes) -> char* {
        char* p = ws + off;
        off += (bytes + 255) & ~(size_t)255;
        return p;
    };
    float*  xf  = (float*)carve((size_t)MM * DD * 4);   // running activations f32
    bf16_t* xb  = (bf16_t*)carve((size_t)MM * DD * 2);  // bf16 mirror (GEMM A)
    float*  yf  = (float*)carve((size_t)MM * DD * 4);   // pre-LN buffer
    float*  Qf  = (float*)carve((size_t)MM * AA * 4);
    float*  Kf  = (float*)carve((size_t)MM * AA * 4);
    float*  vf  = (float*)carve((size_t)MM * AA * 4);
    bf16_t* Vb  = (bf16_t*)carve((size_t)MM * AA * 2);
    bf16_t* hb  = (bf16_t*)carve((size_t)MM * FF * 2);
    bf16_t* wtq = (bf16_t*)carve((size_t)DD * AA * 2);
    bf16_t* wtk = (bf16_t*)carve((size_t)DD * AA * 2);
    bf16_t* wtv = (bf16_t*)carve((size_t)DD * AA * 2);
    bf16_t* wto = (bf16_t*)carve((size_t)AA * DD * 2);
    bf16_t* wt1 = (bf16_t*)carve((size_t)DD * FF * 2);
    bf16_t* wt2 = (bf16_t*)carve((size_t)FF * DD * 2);
    float*  KV  = (float*)carve((size_t)NB * AA * 4);
    float*  Ks  = (float*)carve((size_t)NB * AA * 4);

    // x -> xf (f32 residual) and xb (bf16 operand)
    convert_x_kernel<<<(MM * DD) / 256, 256, 0, stream>>>(x, xf, xb, MM * DD);

    const dim3 gD(DD / TN, MM / TM);   // Nn = 1024 GEMMs
    const dim3 gA(AA / TN, MM / TM);
    const dim3 gF(FF / TN, MM / TM);   // Nn = 4096 (FFN1)

    for (int b = 0; b < BBLK; ++b) {
        const float* Wq_b = Wq + (size_t)b * DD * AA;
        const float* Wk_b = Wk + (size_t)b * DD * AA;
        const float* Wv_b = Wv + (size_t)b * DD * AA;
        const float* Wo_b = Wo + (size_t)b * AA * DD;
        const float* W1_b = Wf1 + (size_t)b * DD * FF;
        const float* W2_b = Wf2 + (size_t)b * FF * DD;

        // Weight convert+transpose to bf16 [Nn][K] (coalesced both sides)
        transpose_w_kernel<<<dim3(AA / 32, DD / 32), 256, 0, stream>>>(Wq_b, wtq, DD, AA);
        transpose_w_kernel<<<dim3(AA / 32, DD / 32), 256, 0, stream>>>(Wk_b, wtk, DD, AA);
        transpose_w_kernel<<<dim3(AA / 32, DD / 32), 256, 0, stream>>>(Wv_b, wtv, DD, AA);
        transpose_w_kernel<<<dim3(DD / 32, AA / 32), 256, 0, stream>>>(Wo_b, wto, AA, DD);
        transpose_w_kernel<<<dim3(FF / 32, DD / 32), 256, 0, stream>>>(W1_b, wt1, DD, FF);
        transpose_w_kernel<<<dim3(DD / 32, FF / 32), 256, 0, stream>>>(W2_b, wt2, FF, DD);

        // Q = elu(x Wq + bq)+1 ; K likewise ; v = x Wv + bv
        gemm_bf16_wmma<1><<<gA, 256, 0, stream>>>(xb, wtq, bq + b * AA, nullptr,
                                                  Qf, nullptr, MM, AA, DD);
        gemm_bf16_wmma<1><<<gA, 256, 0, stream>>>(xb, wtk, bk + b * AA, nullptr,
                                                  Kf, nullptr, MM, AA, DD);
        gemm_bf16_wmma<2><<<gA, 256, 0, stream>>>(xb, wtv, bv + b * AA, nullptr,
                                                  vf, nullptr, MM, AA, DD);

        // KV / Ksum reductions over L
        zero_kernel<<<(2 * NB * AA) / 256, 256, 0, stream>>>(KV, 2 * NB * AA);
        kv_reduce_kernel<<<dim3(NB * (AA / 256), 16), 256, 0, stream>>>(Kf, vf, KV, Ks);

        // V = Q * KV / (Q * Ksum + eps)
        v_map_kernel<<<(int)(((size_t)MM * AA) / 256), 256, 0, stream>>>(Qf, KV, Ks, Vb);

        // attn = gelu(V Wo + bo) + residual -> yf ; then LN1 -> xf, xb
        gemm_bf16_wmma<4><<<gD, 256, 0, stream>>>(Vb, wto, bo + b * DD, xf,
                                                  yf, nullptr, MM, DD, AA);
        ln_kernel<<<MM, 256, 0, stream>>>(yf, ln1s + b * DD, ln1b + b * DD, xf, xb);

        // h = gelu(x Wf1 + bf1) (bf16) ; ff = gelu(h Wf2 + bf2) + residual ; LN2
        gemm_bf16_wmma<3><<<gF, 256, 0, stream>>>(xb, wt1, bf1 + b * FF, nullptr,
                                                  nullptr, hb, MM, FF, DD);
        gemm_bf16_wmma<4><<<gD, 256, 0, stream>>>(hb, wt2, bf2 + b * DD, xf,
                                                  yf, nullptr, MM, DD, FF);

        const bool last = (b == BBLK - 1);
        ln_kernel<<<MM, 256, 0, stream>>>(yf, ln2s + b * DD, ln2b + b * DD,
                                          last ? out : xf, last ? nullptr : xb);
    }
}
